// MLAAttention_6004364279890
// MI455X (gfx1250) — matrix operand order repaired and checked by reading, not compiled, not run on hardware
//
#include <hip/hip_runtime.h>

#define H_    16
#define NOPE_ 128
#define ROPE_ 64
#define VDIM_ 128
#define RANK_ 512
#define QIN_  1536
#define HID_  7168
#define T_    2048
#define DQK_  576                      // RANK + ROPE (concatenated score depth)
#define QF_   (H_ * (NOPE_ + ROPE_))   // 3072
#define SCALE_ 0.07216878364870323f    // (NOPE+ROPE)^-0.5

typedef __attribute__((ext_vector_type(16))) _Float16 v16h;
typedef __attribute__((ext_vector_type(8)))  float    v8f;

// ---------------- WMMA fragment helpers (wave32, ISA 7.12.2 layouts) ----------------
// All loads are aligned 32-bit (one K-pair per dword) to avoid u16 scatter + packing.

// A fragment 16x32 (MxK), f16, row-major source, leading dim lda (even).
// lane 0-15: M=lane, K groups {0-7,16-23}; lane 16-31: M=lane-16, K groups {8-15,24-31}.
__device__ inline v16h load_a_frag(const _Float16* a, int lda) {
  int lane = threadIdx.x & 31;
  int m    = lane & 15;
  int half = lane >> 4;
  union { v16h h; unsigned u[8]; } f;
#pragma unroll
  for (int v = 0; v < 8; ++v) {
    int k  = ((v & 4) ? 16 : 0) + half * 8 + (v & 3) * 2;
    f.u[v] = *(const unsigned*)(a + m * lda + k);
  }
  return f.h;
}

// B fragment 32x16 (KxN) from an N-major source (N rows x K cols, ld even):
// element(k,n) = b[n*ldb + k].  lane%16 = N, lane/16 selects K groups {0-7,16-23} vs {8-15,24-31}.
__device__ inline v16h load_bT_frag(const _Float16* b, int ldb) {
  int lane = threadIdx.x & 31;
  int n    = lane & 15;
  int koff = (lane >> 4) * 8;
  union { v16h h; unsigned u[8]; } f;
#pragma unroll
  for (int v = 0; v < 8; ++v) {
    f.u[v] = *(const unsigned*)(b + n * ldb + ((v & 4) ? 16 : 0) + koff + (v & 3) * 2);
  }
  return f.h;
}

__device__ inline v8f wmma_f16(v16h a, v16h b, v8f c) {
  return __builtin_amdgcn_wmma_f32_16x16x32_f16(false, a, false, b, (short)0, c,
                                                false, false);
}

// ---------------- generic WMMA GEMM: C = A @ B^T, B stored N-major ([N,K] row-major) ---
// block tile 128x128; 8 waves: 4 (M) x 2 (N); wave tile 32x64 -> 2x4 subtiles.
template <bool OUT16>
__global__ __launch_bounds__(256) void gemm_wmma(
    const _Float16* __restrict__ A, int lda, long aoffz,
    const _Float16* __restrict__ B, int ldb, long boffz,
    void* __restrict__ C, int ldc, long coffz, int K, float scale) {
  int wave = threadIdx.x >> 5;
  int wm   = wave & 3;
  int wn   = wave >> 2;
  long z   = blockIdx.z;
  const _Float16* Ab = A + z * aoffz;
  const _Float16* Bb = B + z * boffz;
  int m0 = blockIdx.y * 128 + wm * 32;
  int n0 = blockIdx.x * 128 + wn * 64;

  v8f acc[2][4];
#pragma unroll
  for (int i = 0; i < 2; ++i)
#pragma unroll
    for (int j = 0; j < 4; ++j) acc[i][j] = {};

  for (int k0 = 0; k0 < K; k0 += 32) {
    v16h af0 = load_a_frag(Ab + (long)m0 * lda + k0, lda);
    v16h af1 = load_a_frag(Ab + (long)(m0 + 16) * lda + k0, lda);
#pragma unroll
    for (int j = 0; j < 4; ++j) {
      v16h bf = load_bT_frag(Bb + (long)(n0 + 16 * j) * ldb + k0, ldb);
      acc[0][j] = wmma_f16(af0, bf, acc[0][j]);
      acc[1][j] = wmma_f16(af1, bf, acc[1][j]);
    }
  }

  int lane = threadIdx.x & 31;
  int n    = lane & 15;
  int mb   = (lane >> 4) * 8;
#pragma unroll
  for (int i = 0; i < 2; ++i)
#pragma unroll
    for (int j = 0; j < 4; ++j)
#pragma unroll
      for (int r = 0; r < 8; ++r) {
        long row  = m0 + 16 * i + mb + r;
        long col  = n0 + 16 * j + n;
        float val = acc[i][j][r] * scale;
        if constexpr (OUT16)
          ((_Float16*)C)[z * coffz + row * ldc + col] = (_Float16)val;
        else
          ((float*)C)[z * coffz + row * ldc + col] = val;
      }
}

// ---------------- elementwise kernels ----------------

__global__ void f32_to_f16(const float* __restrict__ in, _Float16* __restrict__ out,
                           long n) {
  long i = (long)blockIdx.x * blockDim.x + threadIdx.x;
  if (i < n) out[i] = (_Float16)in[i];
}

// wukt[h][r][d] = kvb[(h*256+d)*512 + r]  (N-major for the q_lat GEMM)
__global__ void build_wuk_t(const float* __restrict__ kvb, _Float16* __restrict__ wukt) {
  long i = (long)blockIdx.x * blockDim.x + threadIdx.x;
  if (i < (long)H_ * RANK_ * NOPE_) {
    int d = (int)(i % NOPE_);
    int r = (int)((i / NOPE_) % RANK_);
    int h = (int)(i / ((long)NOPE_ * RANK_));
    wukt[i] = (_Float16)kvb[((long)(h * (NOPE_ + VDIM_) + d)) * RANK_ + r];
  }
}

// k_ext[t][0:512] = k_c ; k_ext[t][512:576] = rope_neox(k_pe[t])
__global__ void build_k_ext(const float* __restrict__ kc, const float* __restrict__ kpe,
                            const int* __restrict__ pos, const float* __restrict__ cosc,
                            const float* __restrict__ sinc, _Float16* __restrict__ kext) {
  int t = blockIdx.x;
  int p = pos[t];
  for (int c = threadIdx.x; c < DQK_; c += blockDim.x) {
    float v;
    if (c < RANK_) {
      v = kc[(long)t * RANK_ + c];
    } else {
      int i = c - RANK_;
      if (i < ROPE_ / 2) {
        float x1 = kpe[(long)t * ROPE_ + i];
        float x2 = kpe[(long)t * ROPE_ + ROPE_ / 2 + i];
        v = x1 * cosc[p * (ROPE_ / 2) + i] - x2 * sinc[p * (ROPE_ / 2) + i];
      } else {
        int ii = i - ROPE_ / 2;
        float x1 = kpe[(long)t * ROPE_ + ii];
        float x2 = kpe[(long)t * ROPE_ + ROPE_ / 2 + ii];
        v = x2 * cosc[p * (ROPE_ / 2) + ii] + x1 * sinc[p * (ROPE_ / 2) + ii];
      }
    }
    kext[(long)t * DQK_ + c] = (_Float16)v;
  }
}

// split qf (T x 3072 f32) -> q_nope f16 (T x H x 128), roped+scaled q_pe into
// q_ext[t][h][512:576]
__global__ void split_rope_q(const float* __restrict__ qf, const int* __restrict__ pos,
                             const float* __restrict__ cosc, const float* __restrict__ sinc,
                             _Float16* __restrict__ qnope, _Float16* __restrict__ qext) {
  int t = blockIdx.x;
  int p = pos[t];
  const float* row = qf + (long)t * QF_;
  for (int idx = threadIdx.x; idx < QF_; idx += blockDim.x) {
    int h = idx / (NOPE_ + ROPE_);
    int d = idx % (NOPE_ + ROPE_);
    float v = row[idx];
    if (d < NOPE_) {
      qnope[(long)t * (H_ * NOPE_) + h * NOPE_ + d] = (_Float16)v;
    } else {
      int i = d - NOPE_;
      float out;
      if (i < ROPE_ / 2) {
        float x1 = v;
        float x2 = row[h * (NOPE_ + ROPE_) + NOPE_ + ROPE_ / 2 + i];
        out = x1 * cosc[p * (ROPE_ / 2) + i] - x2 * sinc[p * (ROPE_ / 2) + i];
      } else {
        int ii = i - ROPE_ / 2;
        float x2 = v;
        float x1 = row[h * (NOPE_ + ROPE_) + NOPE_ + ii];
        out = x2 * cosc[p * (ROPE_ / 2) + ii] + x1 * sinc[p * (ROPE_ / 2) + ii];
      }
      qext[(long)t * (H_ * DQK_) + h * DQK_ + RANK_ + i] = (_Float16)(out * SCALE_);
    }
  }
}

// ---------------- flash attention over concatenated 576-dim score space ----------------
// grid: (T/64 query tiles, H). block: 256 threads = 8 waves.
// wave w owns o_lat columns [w*64, w*64+64); online softmax state in LDS.
// V tile is staged transposed in LDS so P@V B-fragments are contiguous dword LDS loads.
#define TQ 64
#define TK 64
#define VT_LD 66   // padded leading dim: 33-dword row stride -> conflict-free reads

__global__ __launch_bounds__(256) void flash_mla(
    const _Float16* __restrict__ qext,  // (T, H, 576), pre-scaled
    const _Float16* __restrict__ kext,  // (T, 576)
    _Float16* __restrict__ olat) {      // (T, H, 512)
  int qt   = blockIdx.x;
  int h    = blockIdx.y;
  int q0   = qt * TQ;
  int wave = threadIdx.x >> 5;
  int lane = threadIdx.x & 31;

  __shared__ _Float16 vt[RANK_][VT_LD];      // 67.6 KB: vt[col][kv-row]
  __shared__ float    s_tile[TQ][TK];        // 16 KB
  __shared__ _Float16 p_tile[TQ][TK];        // 8 KB
  __shared__ float    m_state[TQ];
  __shared__ float    l_state[TQ];
  __shared__ float    alpha_s[TQ];

  if (threadIdx.x < TQ) {
    m_state[threadIdx.x] = -__builtin_inff();
    l_state[threadIdx.x] = 0.f;
  }
  __syncthreads();

  v8f oacc[4][4];
#pragma unroll
  for (int mi = 0; mi < 4; ++mi)
#pragma unroll
    for (int ni = 0; ni < 4; ++ni) oacc[mi][ni] = {};

  const int ldq = H_ * DQK_;
  const _Float16* qbase = qext + (long)q0 * ldq + h * DQK_;

  for (int j = 0; j <= qt; ++j) {
    int s0 = j * TK;

    // ---- phase 0: stage V tile transposed into LDS (coalesced reads) ----
    for (int idx = threadIdx.x; idx < TK * RANK_; idx += 256) {
      int row = idx >> 9;          // kv row within tile (0..63)
      int col = idx & (RANK_ - 1); // latent col (0..511)
      vt[col][row] = kext[(long)(s0 + row) * DQK_ + col];
    }

    // ---- phase 1: S = Qext @ Kext^T : 16 subtiles, 2 per wave ----
    for (int t = wave; t < 16; t += 8) {
      int mi = t >> 2, ni = t & 3;
      v8f acc = {};
      for (int kc = 0; kc < DQK_; kc += 32) {
        v16h af = load_a_frag(qbase + (long)(mi * 16) * ldq + kc, ldq);
        v16h bf = load_bT_frag(kext + (long)(s0 + ni * 16) * DQK_ + kc, DQK_);
        acc = wmma_f16(af, bf, acc);
      }
      int n  = lane & 15;
      int mb = (lane >> 4) * 8;
#pragma unroll
      for (int r = 0; r < 8; ++r) {
        int m    = mb + r;
        int grow = q0 + mi * 16 + m;
        int gcol = s0 + ni * 16 + n;
        s_tile[mi * 16 + m][ni * 16 + n] =
            (gcol <= grow) ? acc[r] : -__builtin_inff();
      }
    }
    __syncthreads();

    // ---- phase 2: online softmax row update (one thread per query row) ----
    if (threadIdx.x < TQ) {
      int m    = threadIdx.x;
      float mx = m_state[m];
      for (int n = 0; n < TK; ++n) mx = fmaxf(mx, s_tile[m][n]);
      float al = __expf(m_state[m] - mx);
      float l  = l_state[m] * al;
      for (int n = 0; n < TK; ++n) {
        float pv = __expf(s_tile[m][n] - mx);
        p_tile[m][n] = (_Float16)pv;
        l += pv;
      }
      m_state[m] = mx;
      l_state[m] = l;
      alpha_s[m] = al;
    }
    __syncthreads();

    // ---- phase 3: rescale accumulators, accumulate P @ V from LDS ----
    {
      int mb = (lane >> 4) * 8;
#pragma unroll
      for (int mi = 0; mi < 4; ++mi)
#pragma unroll
        for (int r = 0; r < 8; ++r) {
          float al = alpha_s[mi * 16 + mb + r];
#pragma unroll
          for (int ni = 0; ni < 4; ++ni) oacc[mi][ni][r] *= al;
        }
#pragma unroll
      for (int kc = 0; kc < TK; kc += 32) {
        v16h af[4];
#pragma unroll
        for (int mi = 0; mi < 4; ++mi)
          af[mi] = load_a_frag(&p_tile[mi * 16][kc], TK);
#pragma unroll
        for (int ni = 0; ni < 4; ++ni) {
          v16h bf = load_bT_frag(&vt[wave * 64 + ni * 16][kc], VT_LD);
#pragma unroll
          for (int mi = 0; mi < 4; ++mi)
            oacc[mi][ni] = wmma_f16(af[mi], bf, oacc[mi][ni]);
        }
      }
    }
    __syncthreads();
  }

  // ---- epilogue: normalize by l, store f16 o_lat ----
  int n  = lane & 15;
  int mb = (lane >> 4) * 8;
#pragma unroll
  for (int mi = 0; mi < 4; ++mi)
#pragma unroll
    for (int ni = 0; ni < 4; ++ni)
#pragma unroll
      for (int r = 0; r < 8; ++r) {
        int m     = mi * 16 + mb + r;
        float val = oacc[mi][ni][r] / l_state[m];
        long row  = q0 + m;
        int col   = wave * 64 + ni * 16 + n;
        olat[row * (H_ * RANK_) + h * RANK_ + col] = (_Float16)val;
      }
}

// ---------------- host launch ----------------

extern "C" void kernel_launch(void* const* d_in, const int* in_sizes, int n_in,
                              void* d_out, int out_size, void* d_ws, size_t ws_size,
                              hipStream_t stream) {
  const float* q    = (const float*)d_in[0];
  const float* kc   = (const float*)d_in[1];
  const float* kpe  = (const float*)d_in[2];
  const int*   pos  = (const int*)d_in[3];
  const float* wq   = (const float*)d_in[4];
  const float* kvb  = (const float*)d_in[5];
  const float* wo   = (const float*)d_in[6];
  const float* cosc = (const float*)d_in[7];
  const float* sinc = (const float*)d_in[8];

  char* ws   = (char*)d_ws;
  size_t cur = 0;
  auto take = [&](size_t bytes) {
    size_t p = cur;
    cur = p + ((bytes + 255) & ~(size_t)255);
    return p;
  };
  _Float16* q16   = (_Float16*)(ws + take((size_t)T_ * QIN_ * 2));
  _Float16* wq16  = (_Float16*)(ws + take((size_t)QF_ * QIN_ * 2));
  _Float16* kvb16 = (_Float16*)(ws + take((size_t)H_ * (NOPE_ + VDIM_) * RANK_ * 2));
  _Float16* wo16  = (_Float16*)(ws + take((size_t)HID_ * (H_ * VDIM_) * 2));
  _Float16* wukt  = (_Float16*)(ws + take((size_t)H_ * RANK_ * NOPE_ * 2));
  _Float16* kext  = (_Float16*)(ws + take((size_t)T_ * DQK_ * 2));
  float*    qf32  = (float*)   (ws + take((size_t)T_ * QF_ * 4));
  _Float16* qnope = (_Float16*)(ws + take((size_t)T_ * H_ * NOPE_ * 2));
  _Float16* qext  = (_Float16*)(ws + take((size_t)T_ * H_ * DQK_ * 2));
  _Float16* olat  = (_Float16*)(ws + take((size_t)T_ * H_ * RANK_ * 2));
  _Float16* o16   = (_Float16*)(ws + take((size_t)T_ * H_ * VDIM_ * 2));

  auto conv = [&](const float* src, _Float16* dst, long n) {
    f32_to_f16<<<dim3((unsigned)((n + 255) / 256)), 256, 0, stream>>>(src, dst, n);
  };
  conv(q, q16, (long)T_ * QIN_);
  conv(wq, wq16, (long)QF_ * QIN_);
  conv(kvb, kvb16, (long)H_ * (NOPE_ + VDIM_) * RANK_);
  conv(wo, wo16, (long)HID_ * (H_ * VDIM_));

  {
    long n = (long)H_ * RANK_ * NOPE_;
    build_wuk_t<<<dim3((unsigned)((n + 255) / 256)), 256, 0, stream>>>(kvb, wukt);
  }
  build_k_ext<<<dim3(T_), 256, 0, stream>>>(kc, kpe, pos, cosc, sinc, kext);

  // qf = q @ wq^T   (2048 x 3072, K=1536)
  gemm_wmma<false><<<dim3(QF_ / 128, T_ / 128, 1), 256, 0, stream>>>(
      q16, QIN_, 0, wq16, QIN_, 0, qf32, QF_, 0, QIN_, 1.0f);

  split_rope_q<<<dim3(T_), 256, 0, stream>>>(qf32, pos, cosc, sinc, qnope, qext);

  // q_lat (per head): qnope_h (2048x128) @ wukt_h^T -> qext[:, h, 0:512] * SCALE
  gemm_wmma<true><<<dim3(RANK_ / 128, T_ / 128, H_), 256, 0, stream>>>(
      qnope, H_ * NOPE_, NOPE_, wukt, NOPE_, (long)RANK_ * NOPE_, qext,
      H_ * DQK_, DQK_, NOPE_, SCALE_);

  // flash attention -> o_lat
  flash_mla<<<dim3(T_ / TQ, H_), 256, 0, stream>>>(qext, kext, olat);

  // o (per head): o_lat_h (2048x512) @ w_uv_h^T (512x128) -> o16[:, h*128:...]
  gemm_wmma<true><<<dim3(1, T_ / 128, H_), 256, 0, stream>>>(
      olat, H_ * RANK_, RANK_, kvb16 + (size_t)NOPE_ * RANK_, RANK_,
      (long)(NOPE_ + VDIM_) * RANK_, o16, H_ * VDIM_, VDIM_, RANK_, 1.0f);

  // out = o @ wo^T  (2048 x 7168, K=2048)
  gemm_wmma<false><<<dim3(HID_ / 128, T_ / 128, 1), 256, 0, stream>>>(
      o16, H_ * VDIM_, 0, wo16, H_ * VDIM_, 0, d_out, HID_, 0, H_ * VDIM_, 1.0f);
}
